// EncoderBlock_71957882077852
// MI455X (gfx1250) — matrix operand order repaired and checked by reading, not compiled, not run on hardware
//
#include <hip/hip_runtime.h>
#include <cstdint>
#include <cstddef>

// ---------------------------------------------------------------------------
// Problem constants (match reference)
// ---------------------------------------------------------------------------
constexpr int Bc   = 2;
constexpr int Sc   = 2048;
constexpr int Ec   = 1024;   // EMB
constexpr int Hc   = 4096;   // HID
constexpr int NHc  = 16;
constexpr int DKc  = 64;
constexpr int Mr   = Bc * Sc;      // 4096 token rows
constexpr float LN_EPS = 1e-6f;

typedef unsigned short u16t;
typedef __attribute__((ext_vector_type(16))) __bf16 v16bf;
typedef __attribute__((ext_vector_type(8)))  float  v8f;
typedef __attribute__((ext_vector_type(4)))  unsigned int u32x4;
typedef __attribute__((ext_vector_type(8)))  int  i32x8;
typedef __attribute__((ext_vector_type(4)))  int  i32x4;

union Frag16 {
    v16bf bf;
    uint4 q[2];
    u16t  us[16];
};

static __device__ __forceinline__ u16t f32_bf16(float f) {
    union { float f; uint32_t u; } v; v.f = f;
    uint32_t r = (v.u + 0x7FFFu + ((v.u >> 16) & 1u)) >> 16;   // RNE
    return (u16t)r;
}

static __device__ __forceinline__ v8f wmma_bf16(v16bf a, v16bf b, v8f c) {
    // D = A(16x32 bf16) * B(32x16 bf16) + C(16x16 f32)
    return __builtin_amdgcn_wmma_f32_16x16x32_bf16(
        false, a, false, b, (short)0, c, false, false);
}

static __device__ __forceinline__ float redmax16(float v) {
    v = fmaxf(v, __shfl_xor(v, 1, 32));
    v = fmaxf(v, __shfl_xor(v, 2, 32));
    v = fmaxf(v, __shfl_xor(v, 4, 32));
    v = fmaxf(v, __shfl_xor(v, 8, 32));
    return v;
}
static __device__ __forceinline__ float redsum16(float v) {
    v += __shfl_xor(v, 1, 32);
    v += __shfl_xor(v, 2, 32);
    v += __shfl_xor(v, 4, 32);
    v += __shfl_xor(v, 8, 32);
    return v;
}

// ---------------------------------------------------------------------------
// Tensor Data Mover support (gfx1250). 2D tile load Global -> LDS with
// hardware row padding (rows of 64B data + 16B pad -> 80B LDS row stride).
// ---------------------------------------------------------------------------
#if __has_builtin(__builtin_amdgcn_tensor_load_to_lds) && \
    __has_builtin(__builtin_amdgcn_s_wait_tensorcnt)
#define USE_TDM 1
#else
#define USE_TDM 0
#endif

#if USE_TDM
static __device__ __forceinline__ void tdm_load_tile_128x32(
        const u16t* gsrc, uint32_t lds_byte_addr, int stride_elems) {
    const uint64_t ga = (uint64_t)(uintptr_t)gsrc;   // byte address, 57-bit
    u32x4 g0;
    g0.x = 1u;                                       // count=1, user descriptor
    g0.y = lds_byte_addr;                            // lds_addr [63:32]
    g0.z = (uint32_t)ga;                             // global_addr low
    g0.w = (uint32_t)((ga >> 32) & 0x01FFFFFFull) | (2u << 30);  // addr hi | type=2

    i32x8 g1;
    // data_size=1 (2B), pad_enable, pad_interval=3 (16 DW = 64B row),
    // pad_amount=3 (4 DW = 16B pad)
    g1[0] = (1 << 16) | (1 << 20) | (3 << 22) | (3 << 25);
    g1[1] = (32 << 16);                 // tensor_dim0[15:0]=32 (in [63:48])
    g1[2] = (128 << 16);                // tensor_dim0 hi=0 | tensor_dim1=128
    g1[3] = (32 << 16);                 // tensor_dim1 hi=0 | tile_dim0=32
    g1[4] = 128;                        // tile_dim1=128 | tile_dim2=0
    g1[5] = stride_elems;               // tensor_dim0_stride low
    g1[6] = 0;                          // stride0 hi | stride1 low
    g1[7] = 0;
    i32x4 z4 = {};
#if __has_include(<hip/amd_detail/amd_gfx1250_TDM.h>)
    i32x8 z8 = {};
    __builtin_amdgcn_tensor_load_to_lds(g0, g1, z4, z4, z8, 0);  // 6-arg toolchain
#else
    __builtin_amdgcn_tensor_load_to_lds(g0, g1, z4, z4, 0);      // 5-arg toolchain
#endif
}
#endif

// ---------------------------------------------------------------------------
// Weight convert + transpose: W[K][N] f32 -> Wt[N][K] bf16  (K-major for WMMA)
// ---------------------------------------------------------------------------
__global__ __launch_bounds__(256)
void cvt_transpose(const float* __restrict__ W, u16t* __restrict__ Wt,
                   int K, int N) {
    int idx = blockIdx.x * 256 + threadIdx.x;
    if (idx >= K * N) return;
    int n = idx / K;
    int k = idx - n * K;
    Wt[idx] = f32_bf16(W[(size_t)k * N + n]);
}

// ---------------------------------------------------------------------------
// LayerNorm (scalar alpha/beta, Bessel-corrected std) -> bf16 output
// ---------------------------------------------------------------------------
__global__ __launch_bounds__(256)
void layernorm_bf16(const float* __restrict__ X, const float* __restrict__ alpha,
                    const float* __restrict__ beta, u16t* __restrict__ Y,
                    int ncols) {
    const int row = blockIdx.x;
    const int tid = threadIdx.x;
    const float* x = X + (size_t)row * ncols;

    float s = 0.f, ss = 0.f;
    for (int c = tid; c < ncols; c += 256) {
        float v = x[c];
        s += v; ss += v * v;
    }
    for (int m = 1; m <= 16; m <<= 1) {
        s  += __shfl_xor(s,  m, 32);
        ss += __shfl_xor(ss, m, 32);
    }
    __shared__ float red[2][8];
    const int wid = tid >> 5;
    if ((tid & 31) == 0) { red[0][wid] = s; red[1][wid] = ss; }
    __syncthreads();
    s = 0.f; ss = 0.f;
    #pragma unroll
    for (int i = 0; i < 8; ++i) { s += red[0][i]; ss += red[1][i]; }

    const float mean = s / (float)ncols;
    float var = (ss - (float)ncols * mean * mean) / (float)(ncols - 1);
    var = fmaxf(var, 0.f);
    const float stdv  = sqrtf(var);
    const float scale = alpha[0] / (stdv + LN_EPS);
    const float bias  = beta[0];
    for (int c = tid; c < ncols; c += 256)
        Y[(size_t)row * ncols + c] = f32_bf16((x[c] - mean) * scale + bias);
}

// ---------------------------------------------------------------------------
// TN GEMM:  C[M,N] = A[M,K](bf16,row) * Bt[N,K](bf16,row)^T  (+bias,relu,res)
// 256 thr = 8 wave32, WG tile 128x128 (waves 4xM,2xN; wave tile 32x64), K step 32
// Tiles staged via TDM DMA (double-buffered) when available.
// ---------------------------------------------------------------------------
template <bool OUT_BF16, bool RELU, bool BIAS, bool RES>
__global__ __launch_bounds__(256)
void gemm_bf16_wmma(const u16t* __restrict__ A, const u16t* __restrict__ Bt,
                    void* __restrict__ Cout, const float* __restrict__ bias,
                    const float* __restrict__ res, int M, int N, int K) {
    // rows padded to 40 elems (80B = 20 banks -> conflict-free fragment reads)
    __shared__ __align__(16) u16t As[2][128 * 40];
    __shared__ __align__(16) u16t Bs[2][128 * 40];

    const int tid  = threadIdx.x;
    const int wid  = tid >> 5;
    const int lane = tid & 31;
    const int half = lane >> 4;
    const int l16  = lane & 15;
    const int wm0  = (wid & 3) * 32;
    const int wn0  = (wid >> 2) * 64;
    const int gm0  = blockIdx.x * 128;
    const int gn0  = blockIdx.y * 128;

    v8f acc[2][4] = {};

#if USE_TDM
    const u16t* Abase = A  + (size_t)gm0 * K;
    const u16t* Bbase = Bt + (size_t)gn0 * K;
    const uint32_t ldsA[2] = { (uint32_t)(uintptr_t)&As[0][0],
                               (uint32_t)(uintptr_t)&As[1][0] };
    const uint32_t ldsB[2] = { (uint32_t)(uintptr_t)&Bs[0][0],
                               (uint32_t)(uintptr_t)&Bs[1][0] };
    if (wid == 0) {                       // prologue: tile 0 -> buffer 0
        tdm_load_tile_128x32(Abase, ldsA[0], K);
        tdm_load_tile_128x32(Bbase, ldsB[0], K);
    }
    int buf = 0;
    for (int k0 = 0; k0 < K; k0 += 32) {
        if (wid == 0) {
            if (k0 + 32 < K) {            // prefetch next tile into other buffer
                tdm_load_tile_128x32(Abase + k0 + 32, ldsA[buf ^ 1], K);
                tdm_load_tile_128x32(Bbase + k0 + 32, ldsB[buf ^ 1], K);
                __builtin_amdgcn_s_wait_tensorcnt(2);   // current tile landed
            } else {
                __builtin_amdgcn_s_wait_tensorcnt(0);
            }
        }
        __syncthreads();
        const u16t* as = As[buf];
        const u16t* bs = Bs[buf];
#else
    const int lrow = tid >> 1;
    const int lseg = (tid & 1) * 16;
    const u16t* Ap = A  + (size_t)(gm0 + lrow) * K + lseg;
    const u16t* Bp = Bt + (size_t)(gn0 + lrow) * K + lseg;
    for (int k0 = 0; k0 < K; k0 += 32) {
        uint4 a0 = *(const uint4*)(Ap + k0);
        uint4 a1 = *(const uint4*)(Ap + k0 + 8);
        uint4 b0 = *(const uint4*)(Bp + k0);
        uint4 b1 = *(const uint4*)(Bp + k0 + 8);
        if (k0 + 32 < K) {
            __builtin_prefetch(Ap + k0 + 32, 0, 1);
            __builtin_prefetch(Bp + k0 + 32, 0, 1);
        }
        *(uint4*)&As[0][lrow * 40 + lseg]     = a0;
        *(uint4*)&As[0][lrow * 40 + lseg + 8] = a1;
        *(uint4*)&Bs[0][lrow * 40 + lseg]     = b0;
        *(uint4*)&Bs[0][lrow * 40 + lseg + 8] = b1;
        __syncthreads();
        const u16t* as = As[0];
        const u16t* bs = Bs[0];
#endif
        Frag16 af[2];
        #pragma unroll
        for (int mt = 0; mt < 2; ++mt) {
            const int am = wm0 + mt * 16 + l16;
            af[mt].q[0] = *(const uint4*)&as[am * 40 + half * 8];       // K = half*8+e
            af[mt].q[1] = *(const uint4*)&as[am * 40 + 16 + half * 8];  // K = 16+half*8+e
        }
        #pragma unroll
        for (int nt = 0; nt < 4; ++nt) {
            Frag16 bf;
            const int bn = wn0 + nt * 16 + l16;
            bf.q[0] = *(const uint4*)&bs[bn * 40 + half * 8];           // K = half*8+e
            bf.q[1] = *(const uint4*)&bs[bn * 40 + 16 + half * 8];
            #pragma unroll
            for (int mt = 0; mt < 2; ++mt)
                acc[mt][nt] = wmma_bf16(af[mt].bf, bf.bf, acc[mt][nt]);
        }
        __syncthreads();
#if USE_TDM
        buf ^= 1;
#endif
    }

    // epilogue: C/D layout -> lane holds row (r + 8*half), col l16
    #pragma unroll
    for (int nt = 0; nt < 4; ++nt) {
        const int col  = gn0 + wn0 + nt * 16 + l16;
        const float bv = BIAS ? bias[col] : 0.f;
        #pragma unroll
        for (int mt = 0; mt < 2; ++mt) {
            #pragma unroll
            for (int r = 0; r < 8; ++r) {
                const int row = gm0 + wm0 + mt * 16 + r + 8 * half;
                float v = acc[mt][nt][r] + bv;
                if (RELU) v = fmaxf(v, 0.f);
                if (RES)  v += res[(size_t)row * N + col];
                if (OUT_BF16) ((u16t*)Cout)[(size_t)row * N + col] = f32_bf16(v);
                else          ((float*)Cout)[(size_t)row * N + col] = v;
            }
        }
    }
}

// ---------------------------------------------------------------------------
// V transpose: V[b][s][h*64+d] bf16 -> Vt[b][h][d][s] bf16
// ---------------------------------------------------------------------------
__global__ __launch_bounds__(256)
void transpose_v(const u16t* __restrict__ V, u16t* __restrict__ Vt) {
    int idx = blockIdx.x * 256 + threadIdx.x;    // total B*NH*DK*S = 4M
    const int s = idx & (Sc - 1);
    int rest = idx >> 11;
    const int d = rest & (DKc - 1); rest >>= 6;
    const int h = rest & (NHc - 1);
    const int b = rest >> 4;
    Vt[idx] = V[((size_t)(b * Sc + s)) * Ec + h * DKc + d];
}

// ---------------------------------------------------------------------------
// Flash attention: one wave32 per 16-row Q tile. 128 thr = 4 waves per WG.
// Q,K bf16 [B*S, E]; Vt bf16 [B,H,DK,S]; mask int [B,S]; O bf16 [B*S, E]
// ---------------------------------------------------------------------------
__global__ __launch_bounds__(128)
void flash_attn(const u16t* __restrict__ Q, const u16t* __restrict__ Kb,
                const u16t* __restrict__ Vt, const int* __restrict__ mask,
                u16t* __restrict__ O) {
    __shared__ __align__(16) u16t pbuf[4][16 * 40];

    const int tid  = threadIdx.x;
    const int wid  = tid >> 5;
    const int lane = tid & 31;
    const int half = lane >> 4;
    const int l16  = lane & 15;

    const int w   = blockIdx.x * 4 + wid;   // 0..4095
    const int b   = w >> 11;
    const int rem = w & 2047;
    const int h   = rem >> 7;
    const int q0  = (rem & 127) * 16;

    Frag16 qf[2];
    #pragma unroll
    for (int c = 0; c < 2; ++c) {
        const u16t* qp = Q + ((size_t)(b * Sc + q0 + l16)) * Ec + h * DKc + c * 32;
        qf[c].q[0] = *(const uint4*)(qp + half * 8);
        qf[c].q[1] = *(const uint4*)(qp + 16 + half * 8);
    }

    v8f acc[4] = {};
    float mrow[8], lrow[8];
    #pragma unroll
    for (int r = 0; r < 8; ++r) { mrow[r] = -3.0e38f; lrow[r] = 0.f; }

    const int* mptr = mask + b * Sc;
    u16t* pb = pbuf[wid];

    for (int kc = 0; kc < Sc; kc += 32) {
        v8f s[2] = {};
        #pragma unroll
        for (int t = 0; t < 2; ++t) {
            const u16t* kp = Kb + ((size_t)(b * Sc + kc + t * 16 + l16)) * Ec + h * DKc;
            Frag16 kb0, kb1;
            kb0.q[0] = *(const uint4*)(kp + half * 8);           // d = half*8+e
            kb0.q[1] = *(const uint4*)(kp + 16 + half * 8);
            kb1.q[0] = *(const uint4*)(kp + 32 + half * 8);      // d = 32+half*8+e
            kb1.q[1] = *(const uint4*)(kp + 48 + half * 8);
            s[t] = wmma_bf16(qf[0].bf, kb0.bf, s[t]);
            s[t] = wmma_bf16(qf[1].bf, kb1.bf, s[t]);
        }
        const bool v0 = mptr[kc + l16] != 0;
        const bool v1 = mptr[kc + 16 + l16] != 0;

        float sm0[8], sm1[8];
        #pragma unroll
        for (int e = 0; e < 8; ++e) {
            sm0[e] = v0 ? s[0][e] * 0.125f : -3.0e38f;
            sm1[e] = v1 ? s[1][e] * 0.125f : -3.0e38f;
        }

        #pragma unroll
        for (int r = 0; r < 8; ++r) {
            float rm = redmax16(fmaxf(sm0[r], sm1[r]));
            const float mn   = fmaxf(mrow[r], rm);
            const float corr = __expf(mrow[r] - mn);
            const float p0   = __expf(sm0[r] - mn);
            const float p1   = __expf(sm1[r] - mn);
            lrow[r] = lrow[r] * corr + redsum16(p0 + p1);
            mrow[r] = mn;
            #pragma unroll
            for (int j = 0; j < 4; ++j) acc[j][r] *= corr;
            pb[(r + 8 * half) * 40 + l16]      = f32_bf16(p0);
            pb[(r + 8 * half) * 40 + 16 + l16] = f32_bf16(p1);
        }
        asm volatile("s_wait_dscnt 0" ::: "memory");

        Frag16 pf;
        pf.q[0] = *(const uint4*)&pb[l16 * 40 + half * 8];
        pf.q[1] = *(const uint4*)&pb[l16 * 40 + 16 + half * 8];
        #pragma unroll
        for (int j = 0; j < 4; ++j) {
            const u16t* vp = Vt + ((size_t)((b * NHc + h) * DKc + j * 16 + l16)) * Sc + kc;
            Frag16 vf;
            vf.q[0] = *(const uint4*)(vp + half * 8);            // key = half*8+e
            vf.q[1] = *(const uint4*)(vp + 16 + half * 8);
            acc[j] = wmma_bf16(pf.bf, vf.bf, acc[j]);
        }
    }

    #pragma unroll
    for (int r = 0; r < 8; ++r) {
        const float inv = lrow[r] > 0.f ? 1.0f / lrow[r] : 0.f;
        const size_t row = (size_t)(b * Sc + q0 + r + 8 * half);
        #pragma unroll
        for (int j = 0; j < 4; ++j)
            O[row * Ec + h * DKc + j * 16 + l16] = f32_bf16(acc[j][r] * inv);
    }
}

// ---------------------------------------------------------------------------
// Host launcher
// ---------------------------------------------------------------------------
extern "C" void kernel_launch(void* const* d_in, const int* in_sizes, int n_in,
                              void* d_out, int out_size, void* d_ws, size_t ws_size,
                              hipStream_t stream) {
    (void)in_sizes; (void)n_in; (void)out_size; (void)ws_size;

    const float* x      = (const float*)d_in[0];
    const int*   mask   = (const int*)  d_in[1];
    const float* wq     = (const float*)d_in[2];
    const float* wk     = (const float*)d_in[3];
    const float* wv     = (const float*)d_in[4];
    const float* wo     = (const float*)d_in[5];
    const float* ff1w   = (const float*)d_in[6];
    const float* ff1b   = (const float*)d_in[7];
    const float* ff2w   = (const float*)d_in[8];
    const float* ff2b   = (const float*)d_in[9];
    const float* ln1a   = (const float*)d_in[10];
    const float* ln1b   = (const float*)d_in[11];
    const float* ln2a   = (const float*)d_in[12];
    const float* ln2b   = (const float*)d_in[13];
    float* out = (float*)d_out;

    char* p = (char*)d_ws;
    auto take = [&](size_t bytes) { char* r = p; p += (bytes + 255) & ~size_t(255); return r; };
    u16t* wq_t  = (u16t*)take((size_t)Ec * Ec * 2);
    u16t* wk_t  = (u16t*)take((size_t)Ec * Ec * 2);
    u16t* wv_t  = (u16t*)take((size_t)Ec * Ec * 2);
    u16t* wo_t  = (u16t*)take((size_t)Ec * Ec * 2);
    u16t* ff1_t = (u16t*)take((size_t)Ec * Hc * 2);
    u16t* ff2_t = (u16t*)take((size_t)Hc * Ec * 2);
    u16t* xln   = (u16t*)take((size_t)Mr * Ec * 2);
    u16t* qb    = (u16t*)take((size_t)Mr * Ec * 2);
    u16t* kb    = (u16t*)take((size_t)Mr * Ec * 2);
    u16t* vb    = (u16t*)take((size_t)Mr * Ec * 2);
    u16t* vt    = (u16t*)take((size_t)Mr * Ec * 2);
    u16t* ab    = (u16t*)take((size_t)Mr * Ec * 2);
    float* hbuf = (float*)take((size_t)Mr * Ec * 4);
    u16t* hln   = (u16t*)take((size_t)Mr * Ec * 2);
    u16t* mid   = (u16t*)take((size_t)Mr * Hc * 2);

    cvt_transpose<<<(Ec * Ec + 255) / 256, 256, 0, stream>>>(wq,   wq_t,  Ec, Ec);
    cvt_transpose<<<(Ec * Ec + 255) / 256, 256, 0, stream>>>(wk,   wk_t,  Ec, Ec);
    cvt_transpose<<<(Ec * Ec + 255) / 256, 256, 0, stream>>>(wv,   wv_t,  Ec, Ec);
    cvt_transpose<<<(Ec * Ec + 255) / 256, 256, 0, stream>>>(wo,   wo_t,  Ec, Ec);
    cvt_transpose<<<(Ec * Hc + 255) / 256, 256, 0, stream>>>(ff1w, ff1_t, Ec, Hc);
    cvt_transpose<<<(Hc * Ec + 255) / 256, 256, 0, stream>>>(ff2w, ff2_t, Hc, Ec);

    layernorm_bf16<<<Mr, 256, 0, stream>>>(x, ln1a, ln1b, xln, Ec);

    dim3 blk(256);
    dim3 gEE(Mr / 128, Ec / 128);
    gemm_bf16_wmma<true, false, false, false><<<gEE, blk, 0, stream>>>(
        xln, wq_t, qb, nullptr, nullptr, Mr, Ec, Ec);
    gemm_bf16_wmma<true, false, false, false><<<gEE, blk, 0, stream>>>(
        xln, wk_t, kb, nullptr, nullptr, Mr, Ec, Ec);
    gemm_bf16_wmma<true, false, false, false><<<gEE, blk, 0, stream>>>(
        xln, wv_t, vb, nullptr, nullptr, Mr, Ec, Ec);

    transpose_v<<<(Mr * Ec) / 256, 256, 0, stream>>>(vb, vt);

    flash_attn<<<(Bc * NHc * (Sc / 16)) / 4, 128, 0, stream>>>(qb, kb, vt, mask, ab);

    gemm_bf16_wmma<false, false, false, true><<<gEE, blk, 0, stream>>>(
        ab, wo_t, hbuf, nullptr, x, Mr, Ec, Ec);

    layernorm_bf16<<<Mr, 256, 0, stream>>>(hbuf, ln2a, ln2b, hln, Ec);

    dim3 gEH(Mr / 128, Hc / 128);
    gemm_bf16_wmma<true, true, true, false><<<gEH, blk, 0, stream>>>(
        hln, ff1_t, mid, ff1b, nullptr, Mr, Hc, Ec);

    gemm_bf16_wmma<false, false, true, true><<<gEE, blk, 0, stream>>>(
        mid, ff2_t, out, ff2b, hbuf, Mr, Ec, Hc);
}
